// modular_Attention_41360535061095
// MI455X (gfx1250) — matrix operand order repaired and checked by reading, not compiled, not run on hardware
//
#include <hip/hip_runtime.h>

#define B_ 4
#define S_ 4096
#define D_ 512
#define U_ 64

typedef __attribute__((ext_vector_type(16))) _Float16 v16h;
typedef __attribute__((ext_vector_type(8)))  _Float16 v8h;
typedef __attribute__((ext_vector_type(8)))  float    v8f;
typedef __attribute__((ext_vector_type(4)))  unsigned int u32x4;
typedef __attribute__((ext_vector_type(8)))  int          i32x8;
typedef __attribute__((ext_vector_type(4)))  int          i32x4;

union V16H { v16h v; v8h h[2]; _Float16 e[16]; };
union V8F  { v8f  v; float e[8]; };
union V8H  { v8h  v; _Float16 e[8]; };

// ---------------------------------------------------------------------------
// Tensor Data Mover: build D# (group0 + group1 per ISA 8.3/8.4) and issue a
// 2D (or 1D when tile1==0) global->LDS DMA. 6-arg builtin form (clang-23).
// ---------------------------------------------------------------------------
__device__ __forceinline__ void tdm_load_to_lds(
    unsigned lds_addr, unsigned long long gaddr,
    unsigned data_size,               // 0=1B,1=2B,2=4B,3=8B
    unsigned tile0, unsigned tile1,   // tile dims in data_size units (tile1=0 -> 1D)
    unsigned long long tdim0, unsigned long long tdim1,
    unsigned long long stride0)       // row stride in data_size units
{
    u32x4 g0;
    g0.x = 1u;                                           // count=1, no gather
    g0.y = lds_addr;                                     // lds_addr [63:32]
    g0.z = (unsigned)gaddr;                              // global_addr [95:64]
    g0.w = (unsigned)((gaddr >> 32) & 0x1FFFFFFull)      // global_addr [120:96]
         | (2u << 30);                                   // type=2 ("image")
    i32x8 g1;
    g1[0] = (int)(data_size << 16);                      // mask=0, flags=0
    g1[1] = (int)((tdim0 & 0xFFFFull) << 16);            // tensor_dim0 lo16
    g1[2] = (int)(((tdim0 >> 16) & 0xFFFFull) | ((tdim1 & 0xFFFFull) << 16));
    g1[3] = (int)(((tdim1 >> 16) & 0xFFFFull) | ((unsigned long long)tile0 << 16));
    g1[4] = (int)(tile1 & 0xFFFFu);                      // tile_dim1, tile_dim2=0
    g1[5] = (int)(stride0 & 0xFFFFFFFFull);              // tensor_dim0_stride lo32
    g1[6] = (int)((stride0 >> 32) & 0xFFFFull);          // stride0 hi16, stride1=0
    g1[7] = 0;
    i32x4 z4 = {0, 0, 0, 0};                             // groups 2/3 unused (2D)
    i32x8 z8 = {0, 0, 0, 0, 0, 0, 0, 0};                 // extra group (6-arg form)
    __builtin_amdgcn_tensor_load_to_lds(g0, g1, z4, z4, z8, 0);
}

// ---------------------------------------------------------------------------
// Prep kernels: one-time fp32 -> f16 conversion (clip fused into x).
// ---------------------------------------------------------------------------
__global__ __launch_bounds__(256) void convert_x_kernel(
    const float* __restrict__ x, _Float16* __restrict__ xh)
{
    size_t i = (size_t)blockIdx.x * 256 + threadIdx.x;   // B*S*D = 8M elements
    float f = x[i];
    xh[i] = (_Float16)fminf(1.0f, fmaxf(-1.0f, f));
}

__global__ __launch_bounds__(256) void convert_w_kernel(
    const float* __restrict__ Wq, const float* __restrict__ Wk,
    const float* __restrict__ Wv, _Float16* __restrict__ Wh)
{
    int i = blockIdx.x * 256 + threadIdx.x;              // D*U = 32768 each
    Wh[i]               = (_Float16)Wq[i];
    Wh[D_ * U_ + i]     = (_Float16)Wk[i];
    Wh[2 * D_ * U_ + i] = (_Float16)Wv[i];
}

// ---------------------------------------------------------------------------
// Projection: one wave per (batch, 16-row tile) computes Q,K,V together so x
// is read once. 12 resident f32 accumulators; pure f16 fragment loads.
//   Q [B,S,U] row major, Kt [B,U,S] transposed, V [B,S,U] row major.
// ---------------------------------------------------------------------------
__global__ __launch_bounds__(32) void proj_qkv_kernel(
    const _Float16* __restrict__ xh,   // [B,S,D] f16 (pre-clipped)
    const _Float16* __restrict__ Wh,   // [3,D,U] f16
    _Float16* __restrict__ Qf,
    _Float16* __restrict__ Ktf,
    _Float16* __restrict__ Vf)
{
    const int lane = threadIdx.x;
    const int h    = lane >> 4;
    const int lm   = lane & 15;
    const int tile = blockIdx.x;                  // 0 .. B*(S/16)-1
    const int b    = tile / (S_ / 16);
    const int m0   = (tile % (S_ / 16)) * 16;

    V8F acc[3][4];
    #pragma unroll
    for (int mat = 0; mat < 3; ++mat)
        #pragma unroll
        for (int nt = 0; nt < 4; ++nt)
            #pragma unroll
            for (int r = 0; r < 8; ++r) acc[mat][nt].e[r] = 0.0f;

    const _Float16* xrow = xh + ((size_t)b * S_ + (m0 + lm)) * D_;

    for (int k0 = 0; k0 < D_; k0 += 32) {
        V16H a;                                    // A frag: two 16B runs
        a.h[0] = *(const v8h*)(xrow + k0 + h * 8);
        a.h[1] = *(const v8h*)(xrow + k0 + 16 + h * 8);
        #pragma unroll
        for (int mat = 0; mat < 3; ++mat) {
            const _Float16* wrow = Wh + mat * D_ * U_ + (size_t)k0 * U_ + lm;
            #pragma unroll
            for (int nt = 0; nt < 4; ++nt) {
                v16h bf;  // lane = N column, elements = K (table order)
                #pragma unroll
                for (int s = 0; s < 16; ++s)
                    bf[s] = wrow[(size_t)(((s & 8) << 1) + 8 * h + (s & 7)) * U_ + nt * 16];
                acc[mat][nt].v = __builtin_amdgcn_wmma_f32_16x16x32_f16(
                    false, a.v, false, bf, (short)0, acc[mat][nt].v, false, false);
            }
        }
    }

    // C layout: acc[mat][nt].e[r] = (M = m0 + r + 8h, N = nt*16 + lm)
    _Float16* qdst = Qf + (size_t)b * S_ * U_;
    _Float16* vdst = Vf + (size_t)b * S_ * U_;
    #pragma unroll
    for (int nt = 0; nt < 4; ++nt) {
        // K: transposed contiguous 16B store per lane
        const int u = nt * 16 + lm;
        V8H pk;
        #pragma unroll
        for (int r = 0; r < 8; ++r) pk.e[r] = (_Float16)acc[1][nt].e[r];
        *(v8h*)(Ktf + ((size_t)b * U_ + u) * S_ + m0 + 8 * h) = pk.v;
        // Q and V: row-major scatter (column per lane)
        #pragma unroll
        for (int r = 0; r < 8; ++r) {
            qdst[(size_t)(m0 + r + 8 * h) * U_ + nt * 16 + lm] = (_Float16)acc[0][nt].e[r];
            vdst[(size_t)(m0 + r + 8 * h) * U_ + nt * 16 + lm] = (_Float16)acc[2][nt].e[r];
        }
    }
}

// ---------------------------------------------------------------------------
// Attention: O[m] = sum_t sigmoid(Q[m]·K[t]) V[t]  (sigmoid -> plain accum).
// 4 waves / workgroup (64 query rows). K/V staged 64 keys at a time into
// double-buffered LDS by the Tensor Data Mover (issued by wave 0, tracked by
// TENSORcnt); WMMA fragments then come from LDS (K/V per element, P as 16B runs).
// ---------------------------------------------------------------------------
__global__ __launch_bounds__(128) void attn_sigmoid_kernel(
    const _Float16* __restrict__ Qf,   // [B,S,U]
    const _Float16* __restrict__ Ktf,  // [B,U,S]
    const _Float16* __restrict__ Vf,   // [B,S,U]
    float* __restrict__ out)           // [B,S,U] fp32
{
    __shared__ _Float16 kvbuf[2][8192]; // per buf: Kt tile 64x64 | V tile 64x64
    __shared__ _Float16 pbuf[4][512];   // per-wave 16x32 P tile

    const int tid  = threadIdx.x;
    const int lane = tid & 31;
    const int wave = tid >> 5;
    const int h    = lane >> 4;
    const int lm   = lane & 15;
    const int b    = blockIdx.x / (S_ / 64);
    const int m0   = (blockIdx.x % (S_ / 64)) * 64 + wave * 16;

    const _Float16* ktb = Ktf + (size_t)b * U_ * S_;
    const _Float16* vb  = Vf  + (size_t)b * S_ * U_;

    // Defined-ness init of DMA-written buffer (TDM overwrites it).
    {
        v8h z = {};
        v8h* p = (v8h*)(&kvbuf[0][0]);
        for (int i = tid; i < 2048; i += 128) p[i] = z;
    }
    __syncthreads();

    // Resident Q A-fragments (2 chunks of K=32 over U=64).
    V16H aq[2];
    const _Float16* qrow = Qf + ((size_t)b * S_ + (m0 + lm)) * U_;
    #pragma unroll
    for (int c = 0; c < 2; ++c) {
        aq[c].h[0] = *(const v8h*)(qrow + c * 32 + h * 8);
        aq[c].h[1] = *(const v8h*)(qrow + c * 32 + 16 + h * 8);
    }

    V8F oacc[4];
    #pragma unroll
    for (int ut = 0; ut < 4; ++ut)
        #pragma unroll
        for (int r = 0; r < 8; ++r) oacc[ut].e[r] = 0.0f;

    const int NT = S_ / 64;   // 64 key stages of 64 keys

    // Stage issue: Kt tile (2D: 64 rows x 128B, stride S*2B) + V tile (1D 8KB)
    auto issue_stage = [&](int st, int buf) {
        unsigned ldsK = (unsigned)(uintptr_t)(&kvbuf[buf][0]);
        unsigned ldsV = (unsigned)(uintptr_t)(&kvbuf[buf][4096]);
        unsigned long long gaK = (unsigned long long)(uintptr_t)ktb
                               + (unsigned long long)st * 64ull * 2ull;
        unsigned long long gaV = (unsigned long long)(uintptr_t)vb
                               + (unsigned long long)st * 64ull * U_ * 2ull;
        // Kt: data_size=4B, tile 32x64 dwords, row stride 2048 dwords
        tdm_load_to_lds(ldsK, gaK, 2u, 32u, 64u,
                        (unsigned long long)(S_ * 2 / 4), 64ull,
                        (unsigned long long)(S_ * 2 / 4));
        // V: data_size=8B, 1D tile of 1024 qwords (contiguous 8KB)
        tdm_load_to_lds(ldsV, gaV, 3u, 1024u, 0u,
                        (unsigned long long)(S_ * U_ * 2 / 8), 0ull, 0ull);
    };

    if (wave == 0) issue_stage(0, 0);

    for (int st = 0; st < NT; ++st) {
        const int buf = st & 1;
        if (wave == 0) {
            if (st + 1 < NT) {
                issue_stage(st + 1, buf ^ 1);
                __builtin_amdgcn_s_wait_tensorcnt(2);  // stage st done (in-order)
            } else {
                __builtin_amdgcn_s_wait_tensorcnt(0);
            }
        }
        __syncthreads();                                // tile visible to all waves

        const _Float16* kvK = &kvbuf[buf][0];           // [u][64 keys]
        const _Float16* kvV = &kvbuf[buf][4096];        // [key][64 u]

        #pragma unroll
        for (int ks = 0; ks < 64; ks += 32) {           // two 32-key subtiles
            // ---- scores: 16 q x 32 keys ----
            #pragma unroll
            for (int nt = 0; nt < 2; ++nt) {
                V8F sacc;
                #pragma unroll
                for (int r = 0; r < 8; ++r) sacc.e[r] = 0.0f;
                #pragma unroll
                for (int c = 0; c < 2; ++c) {
                    v16h bk;
                    #pragma unroll
                    for (int s = 0; s < 16; ++s)
                        bk[s] = kvK[(size_t)(c * 32 + ((s & 8) << 1) + 8 * h + (s & 7)) * 64
                                    + ks + nt * 16 + lm];
                    sacc.v = __builtin_amdgcn_wmma_f32_16x16x32_f16(
                        false, aq[c].v, false, bk, (short)0, sacc.v, false, false);
                }
                #pragma unroll
                for (int r = 0; r < 8; ++r) {
                    float s = sacc.e[r];
                    s = 1.0f / (1.0f + __expf(-s));
                    pbuf[wave][(r + 8 * h) * 32 + nt * 16 + lm] = (_Float16)s;
                }
            }
            // ---- P: C-layout -> A-layout via per-wave LDS (DS is in-order) ----
            V16H ap;
            ap.h[0] = *(const v8h*)(&pbuf[wave][0] + lm * 32 + h * 8);
            ap.h[1] = *(const v8h*)(&pbuf[wave][0] + lm * 32 + 16 + h * 8);
            // ---- O += P · V ----
            #pragma unroll
            for (int ut = 0; ut < 4; ++ut) {
                v16h bv;
                #pragma unroll
                for (int s = 0; s < 16; ++s)
                    bv[s] = kvV[(size_t)(ks + ((s & 8) << 1) + 8 * h + (s & 7)) * 64 + ut * 16 + lm];
                oacc[ut].v = __builtin_amdgcn_wmma_f32_16x16x32_f16(
                    false, ap.v, false, bv, (short)0, oacc[ut].v, false, false);
            }
        }
        __syncthreads();                                // safe to overwrite buf
    }

    float* orow = out + (size_t)b * S_ * U_;
    #pragma unroll
    for (int ut = 0; ut < 4; ++ut)
        #pragma unroll
        for (int r = 0; r < 8; ++r)
            orow[(size_t)(m0 + r + 8 * h) * U_ + ut * 16 + lm] = oacc[ut].e[r];
}

extern "C" void kernel_launch(void* const* d_in, const int* in_sizes, int n_in,
                              void* d_out, int out_size, void* d_ws, size_t ws_size,
                              hipStream_t stream)
{
    const float* x  = (const float*)d_in[0];   // [B,S,D,1]
    const float* Wq = (const float*)d_in[1];   // [D,U]
    const float* Wk = (const float*)d_in[2];   // [D,U]
    const float* Wv = (const float*)d_in[3];   // [D,U]

    const size_t perMat = (size_t)B_ * S_ * U_;          // 1M f16 = 2 MB
    _Float16* Qf  = (_Float16*)d_ws;
    _Float16* Ktf = Qf  + perMat;
    _Float16* Vf  = Ktf + perMat;
    _Float16* xh  = Vf  + perMat;                        // 8M f16 = 16 MB
    _Float16* Wh  = xh  + (size_t)B_ * S_ * D_;          // 3*32K f16 = 192 KB

    convert_x_kernel<<<(B_ * S_ * D_) / 256, 256, 0, stream>>>(x, xh);
    convert_w_kernel<<<(D_ * U_) / 256, 256, 0, stream>>>(Wq, Wk, Wv, Wh);

    proj_qkv_kernel<<<B_ * (S_ / 16), 32, 0, stream>>>(xh, Wh, Qf, Ktf, Vf);

    attn_sigmoid_kernel<<<B_ * (S_ / 64), 128, 0, stream>>>(Qf, Ktf, Vf,
                                                            (float*)d_out);
}
